// EuclideanTransformer_39608188403794
// MI455X (gfx1250) — matrix operand order repaired and checked by reading, not compiled, not run on hardware
//
#include <hip/hip_runtime.h>

// ---------------------------------------------------------------------------
// EuclideanTransformer layer for MI455X (gfx1250, wave32).
// Edge filter MLPs on v_wmma_f32_16x16x32_bf16 with pre-packed bf16 weights
// (B-fragment layout, L2-resident) and an A-fragment-packed LDS relay.
// ---------------------------------------------------------------------------

typedef __attribute__((ext_vector_type(16))) __bf16 v16bf;
typedef __attribute__((ext_vector_type(8)))  float  v8f;

#define FDIM   128
#define HEADS  8
#define DHEAD  16
#define LDEG   4
#define DEDIM  32
#define NRBF   32
#define EVD    16
#define TILE_E 16
#define WPB    4   // waves per block

__device__ __forceinline__ float silu_f(float x) {
    return x / (1.0f + __expf(-x));
}

__device__ __forceinline__ v8f wmma_bf16(v16bf a, v16bf b, v8f c) {
    return __builtin_amdgcn_wmma_f32_16x16x32_bf16(false, a, false, b,
                                                   (short)0, c, false, false);
}

// ---------------------------------------------------------------------------
// Fragment layouts (ISA 7.12.2):
//  A (16x32 bf16): lanes 0-15 hold K {0..7,16..23}, lanes 16-31 K {8..15,24..31};
//                  element j <-> kk = (j&7) + hi*8 + (j>>3)*16
//  B (32x16 bf16): lanes 0-15 hold K {0..7,16..23}, lanes 16-31 K {8..15,24..31}; element j <-> kk as for A
//  C/D (16x16 f32): vgpr r <-> m = r + 8*hi, lane <-> n = lane&15
// ---------------------------------------------------------------------------

// Per-edge rbf A fragment: two contiguous 8-float runs per lane -> 4x float4.
__device__ __forceinline__ v16bf loadA_rbf(const float* __restrict__ p, int lane) {
    const int m = lane & 15, hi = lane >> 4;
    const float* base = p + m * NRBF + hi * 8;
    const float4 x0 = *(const float4*)(base + 0);
    const float4 x1 = *(const float4*)(base + 4);
    const float4 x2 = *(const float4*)(base + 16);
    const float4 x3 = *(const float4*)(base + 20);
    v16bf a;
    a[0]  = (__bf16)x0.x; a[1]  = (__bf16)x0.y; a[2]  = (__bf16)x0.z; a[3]  = (__bf16)x0.w;
    a[4]  = (__bf16)x1.x; a[5]  = (__bf16)x1.y; a[6]  = (__bf16)x1.z; a[7]  = (__bf16)x1.w;
    a[8]  = (__bf16)x2.x; a[9]  = (__bf16)x2.y; a[10] = (__bf16)x2.z; a[11] = (__bf16)x2.w;
    a[12] = (__bf16)x3.x; a[13] = (__bf16)x3.y; a[14] = (__bf16)x3.z; a[15] = (__bf16)x3.w;
    return a;
}

// B fragment from pre-packed bf16 weights: one 32B contiguous load per lane.
__device__ __forceinline__ v16bf loadB_p(const v16bf* __restrict__ p, int lane,
                                         int kchunks, int t, int kt) {
    return p[(t * kchunks + kt) * 32 + lane];
}

// ---------------------------------------------------------------------------
// Kernel 0: pack an f32 [K x 128] weight into bf16 B-fragment layout.
// dst index = ((t*kchunks + kt)*32 + lane)*16 + j ; value = src[kk*128 + n]
// ---------------------------------------------------------------------------
__global__ __launch_bounds__(256) void pack_weight_kernel(
    const float* __restrict__ src, __bf16* __restrict__ dst, int kchunks)
{
    const int tid = blockIdx.x * blockDim.x + threadIdx.x;
    const int total = 8 * kchunks * 512;
    if (tid >= total) return;
    const int j    = tid & 15;
    const int lane = (tid >> 4) & 31;
    const int rest = tid >> 9;
    const int kt   = rest % kchunks;
    const int t    = rest / kchunks;
    const int n  = (lane & 15) + 16 * t;
    const int kk = (j & 7) + (lane >> 4) * 8 + (j >> 3) * 16 + kt * 32;
    dst[tid] = (__bf16)src[kk * FDIM + n];
}

// ---------------------------------------------------------------------------
// Kernel 1: per-node q/k/v projections.  q[n,h,:] = x[n,h,:] @ W[h]
// ---------------------------------------------------------------------------
__global__ __launch_bounds__(128) void proj_kernel(
    const float* __restrict__ x,
    const float* __restrict__ Wq, const float* __restrict__ Wk, const float* __restrict__ Wv,
    const float* __restrict__ Wqe, const float* __restrict__ Wke,
    float* __restrict__ Qi, float* __restrict__ Ki, float* __restrict__ Vi,
    float* __restrict__ Qe, float* __restrict__ Ke)
{
    const int n = blockIdx.x, f = threadIdx.x;
    __shared__ float xs[FDIM];
    xs[f] = x[(size_t)n * FDIM + f];
    __syncthreads();

    const int h = f >> 4, eo = f & 15;
    float q = 0.f, k = 0.f, v = 0.f;
#pragma unroll
    for (int d = 0; d < DHEAD; ++d) {
        const float xv = xs[h * DHEAD + d];
        const int wi = h * (DHEAD * DHEAD) + d * DHEAD + eo;
        q += xv * Wq[wi]; k += xv * Wk[wi]; v += xv * Wv[wi];
    }
    const size_t o = (size_t)n * FDIM + f;
    Qi[o] = q; Ki[o] = k; Vi[o] = v;

    const int l = f >> 5, ee = f & 31;
    float qe = 0.f, ke = 0.f;
#pragma unroll
    for (int d = 0; d < DEDIM; ++d) {
        const float xv = xs[l * DEDIM + d];
        const int wi = l * (DEDIM * DEDIM) + d * DEDIM + ee;
        qe += xv * Wqe[wi]; ke += xv * Wke[wi];
    }
    Qe[o] = qe; Ke[o] = ke;
}

// ---------------------------------------------------------------------------
// Kernel 2: per-edge filter MLPs (WMMA) + attention + atomic segment_sum.
// One wave handles 16 edges.  No block-level barriers (per-wave LDS only).
// ---------------------------------------------------------------------------
__global__ __launch_bounds__(128) void edge_kernel(
    const float* __restrict__ rbf,  const float* __restrict__ ev,
    const float* __restrict__ shv,  const float* __restrict__ cut,
    const int*   __restrict__ snd,  const int*   __restrict__ rcv,
    const v16bf* __restrict__ pfi_w1, const float* __restrict__ fi_b1,
    const v16bf* __restrict__ pfi_w2, const float* __restrict__ fi_b2,
    const float* __restrict__ fi_ew1, const float* __restrict__ fi_eb1,
    const v16bf* __restrict__ pfi_ew2, const float* __restrict__ fi_eb2,
    const v16bf* __restrict__ pfe_w1, const float* __restrict__ fe_b1,
    const v16bf* __restrict__ pfe_w2, const float* __restrict__ fe_b2,
    const float* __restrict__ fe_ew1, const float* __restrict__ fe_eb1,
    const v16bf* __restrict__ pfe_ew2, const float* __restrict__ fe_eb2,
    const float* __restrict__ Qi, const float* __restrict__ Ki, const float* __restrict__ Vi,
    const float* __restrict__ Qe, const float* __restrict__ Ke,
    float* __restrict__ att_inv, float* __restrict__ att_ev)
{
    // h1 relay stored directly in A-fragment-packed order: [kchunk][lane][j]
    __shared__ v16bf  sh_h1[WPB][4 * 32];
    __shared__ int    sm_snd[WPB][TILE_E];
    __shared__ int    sm_rcv[WPB][TILE_E];
    __shared__ float  sm_cut[WPB][TILE_E];
    __shared__ float  sm_evi[WPB][TILE_E][LDEG];   // l0 contraction of ev_diff
    __shared__ float  sm_aev[WPB][TILE_E][LDEG];   // alpha_ev

    const int lane = threadIdx.x & 31;
    const int w    = threadIdx.x >> 5;
    const int e0   = (blockIdx.x * WPB + w) * TILE_E;
    const int m16  = lane & 15;
    const int half = lane >> 4;

    // --- edge metadata + ev_diff l0-contraction (lanes 0..15 of each wave) ---
    if (half == 0) {
        const int e = e0 + m16;
        const int s_ = snd[e], r_ = rcv[e];
        sm_snd[w][m16] = s_;
        sm_rcv[w][m16] = r_;
        sm_cut[w][m16] = cut[e];
        float seg[LDEG] = {0.f, 0.f, 0.f, 0.f};
#pragma unroll
        for (int c = 0; c < EVD; ++c) {
            const float d = ev[s_ * EVD + c] - ev[r_ * EVD + c];
            const int l = (c == 0) ? 0 : (c < 4) ? 1 : (c < 9) ? 2 : 3;
            seg[l] += d * d;
        }
#pragma unroll
        for (int l = 0; l < LDEG; ++l) sm_evi[w][m16][l] = seg[l];
    }
    // same-wave LDS ops are in order -> no barrier needed

    const v16bf a_rbf = loadA_rbf(rbf + (size_t)e0 * NRBF, lane);
    __bf16* h1b = (__bf16*)&sh_h1[w][0];

    // ev branch layer 1 (K=4) computed directly into A-fragment layout, K=32 exact
    auto build_ae = [&](const float* ew1, const float* eb1) {
        v16bf a;
#pragma unroll
        for (int j = 0; j < 16; ++j) {
            const int kk = (j & 7) + half * 8 + ((j >> 3) << 4);
            float s = eb1[kk];
#pragma unroll
            for (int d = 0; d < LDEG; ++d)
                s += sm_evi[w][m16][d] * ew1[d * DEDIM + kk];
            a[j] = (__bf16)silu_f(s);
        }
        return a;
    };

    // rbf layer 1: h1 = silu(rbf @ w1 + b1), scattered into LDS in A-packed order
    auto layer1 = [&](const v16bf* pw1, const float* b1) {
#pragma unroll
        for (int t = 0; t < 8; ++t) {
            v8f c = {};
            c = wmma_bf16(a_rbf, loadB_p(pw1, lane, 1, t, 0), c);
            const int n = m16 + t * 16;            // layer-2 K index of this column
            const float bias = b1[n];
            // destination A-fragment coordinates for kk_total = n
            const int kt  = n >> 5;
            const int kk5 = n & 31;
            const int rhi = (kk5 >> 3) & 1;
            const int j   = (kk5 & 7) | (((kk5 >> 4) & 1) << 3);
#pragma unroll
            for (int r = 0; r < 8; ++r) {
                const int m = r + 8 * half;        // edge row
                const float x = silu_f(c[r] + bias);
                h1b[((kt * 32) + m + 16 * rhi) * 16 + j] = (__bf16)x;
            }
        }
    };

    // one 16-column tile of the filter output: (h1 @ w2) + (a_e @ ew2) + biases
    auto tileC = [&](int t, const v16bf* pw2, const float* b2,
                     v16bf a_e, const v16bf* pew2, const float* eb2) {
        v8f c = {};
#pragma unroll
        for (int kt = 0; kt < 4; ++kt)
            c = wmma_bf16(sh_h1[w][kt * 32 + lane],
                          loadB_p(pw2, lane, 4, t, kt), c);
        c = wmma_bf16(a_e, loadB_p(pew2, lane, 1, t, 0), c);
        const int n = m16 + t * 16;
        const float bias = b2[n] + eb2[n];
        v8f o;
#pragma unroll
        for (int r = 0; r < 8; ++r) o[r] = c[r] + bias;
        return o;
    };

    // ================= fi: invariant-channel filter + attention ==============
    layer1(pfi_w1, fi_b1);
    {
        const v16bf ae = build_ae(fi_ew1, fi_eb1);
        for (int t = 0; t < 8; ++t) {               // tile t == head h
            const v8f fw = tileC(t, pfi_w2, fi_b2, ae, pfi_ew2, fi_eb2);
            const int f = t * 16 + m16;             // lane's feature column
#pragma unroll
            for (int r = 0; r < 8; ++r) {
                const int m  = r + 8 * half;        // edge within tile
                const int s_ = sm_snd[w][m], r_ = sm_rcv[w][m];
                float part = Qi[r_ * FDIM + f] * Ki[s_ * FDIM + f] * fw[r];
                part += __shfl_xor(part, 1);
                part += __shfl_xor(part, 2);
                part += __shfl_xor(part, 4);
                part += __shfl_xor(part, 8);        // reduce over d within head
                const float alpha = part * 0.25f;   // / sqrt(DH=16)
                const float v = Vi[s_ * FDIM + f];
                atomicAdd(&att_inv[r_ * FDIM + f], sm_cut[w][m] * alpha * v);
            }
        }
    }

    // ================= fe: equivariant-channel filter + attention ============
    layer1(pfe_w1, fe_b1);                          // reuse LDS h1 (same wave, in order)
    {
        const v16bf ae = build_ae(fe_ew1, fe_eb1);
        for (int l = 0; l < LDEG; ++l) {            // degree l spans tiles 2l, 2l+1
            const v8f ca = tileC(2 * l,     pfe_w2, fe_b2, ae, pfe_ew2, fe_eb2);
            const v8f cb = tileC(2 * l + 1, pfe_w2, fe_b2, ae, pfe_ew2, fe_eb2);
            const int fa = l * DEDIM + m16, fb = fa + 16;
#pragma unroll
            for (int r = 0; r < 8; ++r) {
                const int m  = r + 8 * half;
                const int s_ = sm_snd[w][m], r_ = sm_rcv[w][m];
                float part = Qe[r_ * FDIM + fa] * Ke[s_ * FDIM + fa] * ca[r]
                           + Qe[r_ * FDIM + fb] * Ke[s_ * FDIM + fb] * cb[r];
                part += __shfl_xor(part, 1);
                part += __shfl_xor(part, 2);
                part += __shfl_xor(part, 4);
                part += __shfl_xor(part, 8);
                const float aev = part * 0.17677669529663689f;  // / sqrt(DE=32)
                if (m16 == 0) sm_aev[w][m][l] = aev;
            }
        }
    }

    // scatter equivariant part: lane handles edge m16, components [8*half, 8*half+8)
    {
        const int m = m16, e = e0 + m;
        const int r_ = sm_rcv[w][m];
        const float cu = sm_cut[w][m];
#pragma unroll
        for (int ci = 0; ci < 8; ++ci) {
            const int c = half * 8 + ci;
            const int l = (c == 0) ? 0 : (c < 4) ? 1 : (c < 9) ? 2 : 3;
            const float a = sm_aev[w][m][l];
            atomicAdd(&att_ev[r_ * EVD + c], cu * a * shv[(size_t)e * EVD + c]);
        }
    }
}

// ---------------------------------------------------------------------------
// Kernel 3: residuals, l0-contraction, 132x132 interaction, outputs.
// ---------------------------------------------------------------------------
__global__ __launch_bounds__(128) void final_kernel(
    const float* __restrict__ inv, const float* __restrict__ ev,
    const float* __restrict__ att_inv, const float* __restrict__ att_ev,
    const float* __restrict__ W, const float* __restrict__ b,
    float* __restrict__ out_inv, float* __restrict__ out_ev)
{
    const int n = blockIdx.x, tid = threadIdx.x;
    __shared__ float xs[FDIM + LDEG];
    __shared__ float ev1s[EVD];
    __shared__ float tb[LDEG];

    const float inv1 = inv[(size_t)n * FDIM + tid] + att_inv[(size_t)n * FDIM + tid];
    xs[tid] = inv1;
    if (tid < EVD) ev1s[tid] = ev[(size_t)n * EVD + tid] + att_ev[(size_t)n * EVD + tid];
    __syncthreads();

    if (tid < LDEG) {
        const int segs[5] = {0, 1, 4, 9, 16};
        float s = 0.f;
        for (int c = segs[tid]; c < segs[tid + 1]; ++c) s += ev1s[c] * ev1s[c];
        xs[FDIM + tid] = s;
    }
    __syncthreads();

    float acc = b[tid];
#pragma unroll 4
    for (int c = 0; c < FDIM + LDEG; ++c) acc += xs[c] * W[c * (FDIM + LDEG) + tid];
    out_inv[(size_t)n * FDIM + tid] = inv1 + acc;

    if (tid < LDEG) {
        float a2 = b[FDIM + tid];
        for (int c = 0; c < FDIM + LDEG; ++c) a2 += xs[c] * W[c * (FDIM + LDEG) + FDIM + tid];
        tb[tid] = a2;
    }
    __syncthreads();
    if (tid < EVD) {
        const int l = (tid == 0) ? 0 : (tid < 4) ? 1 : (tid < 9) ? 2 : 3;
        out_ev[(size_t)n * EVD + tid] = ev1s[tid] * (1.f + tb[l]);
    }
}

// ---------------------------------------------------------------------------
extern "C" void kernel_launch(void* const* d_in, const int* in_sizes, int n_in,
                              void* d_out, int out_size, void* d_ws, size_t ws_size,
                              hipStream_t stream) {
    (void)n_in; (void)out_size;
    const float* inv = (const float*)d_in[0];
    const float* ev  = (const float*)d_in[1];
    const float* rbf = (const float*)d_in[2];
    const float* shv = (const float*)d_in[3];
    const float* cut = (const float*)d_in[4];
    const int*   snd = (const int*)d_in[5];
    const int*   rcv = (const int*)d_in[6];

    const float* fi_w1  = (const float*)d_in[7];
    const float* fi_b1  = (const float*)d_in[8];
    const float* fi_w2  = (const float*)d_in[9];
    const float* fi_b2  = (const float*)d_in[10];
    const float* fi_ew1 = (const float*)d_in[11];
    const float* fi_eb1 = (const float*)d_in[12];
    const float* fi_ew2 = (const float*)d_in[13];
    const float* fi_eb2 = (const float*)d_in[14];
    const float* fe_w1  = (const float*)d_in[15];
    const float* fe_b1  = (const float*)d_in[16];
    const float* fe_w2  = (const float*)d_in[17];
    const float* fe_b2  = (const float*)d_in[18];
    const float* fe_ew1 = (const float*)d_in[19];
    const float* fe_eb1 = (const float*)d_in[20];
    const float* fe_ew2 = (const float*)d_in[21];
    const float* fe_eb2 = (const float*)d_in[22];

    const float* Wq_inv = (const float*)d_in[23];
    const float* Wk_inv = (const float*)d_in[24];
    const float* Wv_inv = (const float*)d_in[25];
    const float* Wq_ev  = (const float*)d_in[26];
    const float* Wk_ev  = (const float*)d_in[27];
    const float* int_w  = (const float*)d_in[28];
    const float* int_b  = (const float*)d_in[29];

    const int N = in_sizes[0] / FDIM;     // 16384
    const int E = in_sizes[5];            // 262144

    // workspace layout: 5 projection tables + 2 accumulators (f32), then
    // 6 pre-packed bf16 weight images (B-fragment layout).
    const size_t NF = (size_t)N * FDIM;
    float* ws = (float*)d_ws;
    float* Qi = ws;
    float* Ki = Qi + NF;
    float* Vi = Ki + NF;
    float* Qe = Vi + NF;
    float* Ke = Qe + NF;
    float* att_inv = Ke + NF;
    float* att_ev  = att_inv + NF;

    __bf16* pk = (__bf16*)(att_ev + (size_t)N * EVD);
    const size_t SZ1 = 8 * 1 * 512;   // K=32  weights: 4096 bf16
    const size_t SZ4 = 8 * 4 * 512;   // K=128 weights: 16384 bf16
    __bf16* pfi_w1  = pk;
    __bf16* pfi_w2  = pfi_w1 + SZ1;
    __bf16* pfi_ew2 = pfi_w2 + SZ4;
    __bf16* pfe_w1  = pfi_ew2 + SZ1;
    __bf16* pfe_w2  = pfe_w1 + SZ1;
    __bf16* pfe_ew2 = pfe_w2 + SZ4;
    const size_t need = (6 * NF + (size_t)N * EVD) * sizeof(float)
                      + (4 * SZ1 + 2 * SZ4) * sizeof(__bf16);
    if (ws_size < need) return;

    // zero the segment_sum accumulators (contiguous, capturable async memset)
    hipMemsetAsync(att_inv, 0, (NF + (size_t)N * EVD) * sizeof(float), stream);

    // pack the six filter weight matrices into bf16 B-fragment layout
    pack_weight_kernel<<<(int)(SZ1 + 255) / 256, 256, 0, stream>>>(fi_w1,  pfi_w1,  1);
    pack_weight_kernel<<<(int)(SZ4 + 255) / 256, 256, 0, stream>>>(fi_w2,  pfi_w2,  4);
    pack_weight_kernel<<<(int)(SZ1 + 255) / 256, 256, 0, stream>>>(fi_ew2, pfi_ew2, 1);
    pack_weight_kernel<<<(int)(SZ1 + 255) / 256, 256, 0, stream>>>(fe_w1,  pfe_w1,  1);
    pack_weight_kernel<<<(int)(SZ4 + 255) / 256, 256, 0, stream>>>(fe_w2,  pfe_w2,  4);
    pack_weight_kernel<<<(int)(SZ1 + 255) / 256, 256, 0, stream>>>(fe_ew2, pfe_ew2, 1);

    proj_kernel<<<N, 128, 0, stream>>>(inv, Wq_inv, Wk_inv, Wv_inv, Wq_ev, Wk_ev,
                                       Qi, Ki, Vi, Qe, Ke);

    edge_kernel<<<E / (TILE_E * WPB), 128, 0, stream>>>(
        rbf, ev, shv, cut, snd, rcv,
        (const v16bf*)pfi_w1, fi_b1, (const v16bf*)pfi_w2, fi_b2,
        fi_ew1, fi_eb1, (const v16bf*)pfi_ew2, fi_eb2,
        (const v16bf*)pfe_w1, fe_b1, (const v16bf*)pfe_w2, fe_b2,
        fe_ew1, fe_eb1, (const v16bf*)pfe_ew2, fe_eb2,
        Qi, Ki, Vi, Qe, Ke, att_inv, att_ev);

    final_kernel<<<N, 128, 0, stream>>>(inv, ev, att_inv, att_ev, int_w, int_b,
                                        (float*)d_out, (float*)d_out + NF);
}
